// Attention_82145544503908
// MI455X (gfx1250) — matrix operand order repaired and checked by reading, not compiled, not run on hardware
//
#include <hip/hip_runtime.h>
#include <hip/hip_bf16.h>

#define SEQ  2312
#define LPAD 2336            // SEQ rounded up to multiple of 32
#define EMB  1024
#define NHEAD 16
#define HDIM 64
#define NQKV 3072
#define ROPE_PREFIX 8
#define QSCALE 0.125f        // 64^-0.5

typedef _Float16 v8h  __attribute__((ext_vector_type(8)));
typedef _Float16 v16h __attribute__((ext_vector_type(16)));
typedef float    v8f  __attribute__((ext_vector_type(8)));

// A fragment (16x32 f16, MxK): lane<16 -> row=lane, K 0..15; lane>=16 -> K 16..31
__device__ __forceinline__ v16h load_a_frag(const _Float16* p) {
    union { v16h v; v8h h[2]; } u;
    u.h[0] = *(const v8h*)(p);
    u.h[1] = *(const v8h*)(p + 8);
    return u.v;
}
// B fragment (32x16 f16, KxN) from row-major B^T: lane<16 -> col=lane, K 0..15; lane>=16 -> K 16..31
__device__ __forceinline__ v16h load_b_frag(const _Float16* p) {
    union { v16h v; v8h h[2]; } u;
    u.h[0] = *(const v8h*)(p);
    u.h[1] = *(const v8h*)(p + 8);
    return u.v;
}
#define WMMA_F16(a, b, c) \
    __builtin_amdgcn_wmma_f32_16x16x32_f16(false, (a), false, (b), (short)0, (c), false, false)

// ---------------------------------------------------------------- convert
__global__ void cvt_f32_to_f16(const float* __restrict__ src, _Float16* __restrict__ dst,
                               int n_src, int n_total) {
    int i = blockIdx.x * blockDim.x + threadIdx.x;
    if (i < n_total) dst[i] = (i < n_src) ? (_Float16)src[i] : (_Float16)0.f;
}

// 16x64 GEMM tile, K=EMB. Fully unrolled: straight-line code has no loop phis,
// so every fragment is single-def/single-use -> no register rotation copies;
// the scheduler hoists load clauses ahead of WMMAs up to the pressure limit.
__device__ __forceinline__ void gemm_tile_16x64(const _Float16* __restrict__ ap0,
                                                const _Float16* __restrict__ bp0,
                                                v8f acc[4]) {
#pragma unroll
    for (int k = 0; k < EMB; k += 32) {
        v16h a = load_a_frag(ap0 + k);
#pragma unroll
        for (int t = 0; t < 4; ++t) {
            v16h b = load_b_frag(bp0 + (size_t)t * 16 * EMB + k);
            acc[t] = WMMA_F16(a, b, acc[t]);
        }
    }
}

// ---------------------------------------------------------------- QKV GEMM + bias + RoPE + pack
// wave computes 16(M) x 64(N) of qkv = x @ qkv_w^T; scatters into Q[h][l][d], K[h][l][d], Vt[h][d][l]
__global__ __launch_bounds__(128) void qkv_rope_kernel(
    const _Float16* __restrict__ xh, const _Float16* __restrict__ wh,
    const float* __restrict__ bias, const float* __restrict__ cosT, const float* __restrict__ sinT,
    _Float16* __restrict__ Qh, _Float16* __restrict__ Kh, _Float16* __restrict__ Vt) {
    int wid   = blockIdx.x * 4 + (threadIdx.x >> 5);
    int mtile = wid / 48;
    int nt    = wid - mtile * 48;
    int lane  = threadIdx.x & 31;
    int col   = lane & 15;
    int half  = lane >> 4;
    int m0 = mtile * 16, n0 = nt * 64;

    v8f acc[4] = {};
    gemm_tile_16x64(xh + (size_t)(m0 + col) * EMB + half * 16,
                    wh + (size_t)(n0 + col) * EMB + half * 16, acc);

    int which = n0 >> 10;            // 0=q 1=k 2=v
    int head  = (n0 & 1023) >> 6;
    float bv[4];
#pragma unroll
    for (int t = 0; t < 4; ++t) bv[t] = bias[n0 + t * 16 + col];

#pragma unroll
    for (int j = 0; j < 8; ++j) {
        int row = m0 + j + half * 8;
        bool vrow = row < SEQ;
        float vals[4];
#pragma unroll
        for (int t = 0; t < 4; ++t) vals[t] = acc[t][j] + bv[t];
        if (which < 2 && row >= ROPE_PREFIX && vrow) {
            int r = row - ROPE_PREFIX;
            float tmp[4] = {vals[0], vals[1], vals[2], vals[3]};
#pragma unroll
            for (int t = 0; t < 4; ++t) {
                int d = t * 16 + col;
                float c = cosT[r * HDIM + d];
                float s = sinT[r * HDIM + d];
                float partner = tmp[t ^ 2];             // element at d ^ 32
                vals[t] = tmp[t] * c + (t < 2 ? -partner : partner) * s;
            }
        }
#pragma unroll
        for (int t = 0; t < 4; ++t) {
            int d = t * 16 + col;
            if (which == 0)
                Qh[((size_t)head * LPAD + row) * HDIM + d] = vrow ? (_Float16)(vals[t] * QSCALE) : (_Float16)0.f;
            else if (which == 1)
                Kh[((size_t)head * LPAD + row) * HDIM + d] = vrow ? (_Float16)vals[t] : (_Float16)0.f;
            else
                Vt[((size_t)head * HDIM + d) * LPAD + row] = vrow ? (_Float16)vals[t] : (_Float16)0.f;
        }
    }
}

// ---------------------------------------------------------------- flash attention (1 wave = 1 head x 16 queries)
__global__ __launch_bounds__(128) void attn_kernel(
    const _Float16* __restrict__ Qh, const _Float16* __restrict__ Kh,
    const _Float16* __restrict__ Vt, const unsigned char* __restrict__ maskb,
    _Float16* __restrict__ ctxh) {
    // per-wave private LDS slice; same-wave LDS ops are in-order (DScnt), so no
    // block barrier is needed between the P store and the transposed reload.
    __shared__ __align__(16) _Float16 pb[4][16 * 32];
    int warp  = threadIdx.x >> 5;
    int wid   = blockIdx.x * 4 + warp;
    int head  = wid / (LPAD / 16);
    int qtile = wid - head * (LPAD / 16);
    int lane  = threadIdx.x & 31;
    int col   = lane & 15;
    int half  = lane >> 4;

    const _Float16* qp = Qh + ((size_t)head * LPAD + qtile * 16 + col) * HDIM + half * 16;
    v16h a_lo = load_a_frag(qp);        // d 0..31
    v16h a_hi = load_a_frag(qp + 32);   // d 32..63

    float mrow[8], lrow[8];
    v8f O[4] = {};
#pragma unroll
    for (int j = 0; j < 8; ++j) { mrow[j] = -3.0e38f; lrow[j] = 0.f; }
    _Float16* myp = pb[warp];

    for (int kt = 0; kt < LPAD; kt += 32) {
        const _Float16* kbase = Kh + ((size_t)head * LPAD + kt) * HDIM;
        // K-tile fragments for scores
        v16h bk0 = load_b_frag(kbase + (size_t)col * HDIM + half * 16);
        v16h bk1 = load_b_frag(kbase + (size_t)col * HDIM + 32 + half * 16);
        v16h bk2 = load_b_frag(kbase + (size_t)(col + 16) * HDIM + half * 16);
        v16h bk3 = load_b_frag(kbase + (size_t)(col + 16) * HDIM + 32 + half * 16);
        // V-tile fragments hoisted: independent of softmax, overlap with VALU below
        v16h vb[4];
#pragma unroll
        for (int t = 0; t < 4; ++t)
            vb[t] = load_b_frag(Vt + ((size_t)head * HDIM + t * 16 + col) * LPAD + kt + half * 16);

        v8f s0 = {}, s1 = {};
        s0 = WMMA_F16(a_lo, bk0, s0);
        s0 = WMMA_F16(a_hi, bk1, s0);
        s1 = WMMA_F16(a_lo, bk2, s1);
        s1 = WMMA_F16(a_hi, bk3, s1);

        int key0 = kt + col, key1 = kt + 16 + col;
        bool ok0 = (key0 < SEQ) && (maskb[key0] != 0);
        bool ok1 = (key1 < SEQ) && (maskb[key1] != 0);

        float t0[8], t1[8], tm[8];
#pragma unroll
        for (int j = 0; j < 8; ++j) {
            t0[j] = ok0 ? s0[j] : -1.0e30f;
            t1[j] = ok1 ? s1[j] : -1.0e30f;
            tm[j] = fmaxf(t0[j], t1[j]);
        }
#pragma unroll
        for (int off = 1; off < 16; off <<= 1)
#pragma unroll
            for (int j = 0; j < 8; ++j)
                tm[j] = fmaxf(tm[j], __shfl_xor(tm[j], off, 32));

        float alpha[8], rs[8], p0[8], p1[8];
#pragma unroll
        for (int j = 0; j < 8; ++j) {
            float nm = fmaxf(mrow[j], tm[j]);
            alpha[j] = __expf(mrow[j] - nm);
            mrow[j]  = nm;
            p0[j] = __expf(t0[j] - nm);
            p1[j] = __expf(t1[j] - nm);
            rs[j] = p0[j] + p1[j];
        }
#pragma unroll
        for (int off = 1; off < 16; off <<= 1)
#pragma unroll
            for (int j = 0; j < 8; ++j)
                rs[j] += __shfl_xor(rs[j], off, 32);
#pragma unroll
        for (int j = 0; j < 8; ++j) lrow[j] = lrow[j] * alpha[j] + rs[j];
#pragma unroll
        for (int t = 0; t < 4; ++t)
#pragma unroll
            for (int j = 0; j < 8; ++j) O[t][j] *= alpha[j];

        // C-layout probs -> LDS -> A-fragment layout (same-wave, in-order LDS)
#pragma unroll
        for (int j = 0; j < 8; ++j) {
            myp[(j + half * 8) * 32 + col]      = (_Float16)p0[j];
            myp[(j + half * 8) * 32 + 16 + col] = (_Float16)p1[j];
        }
        v16h pa;
        {
            union { v16h v; v8h h[2]; } u;
            const _Float16* pr = myp + col * 32 + half * 16;
            u.h[0] = *(const v8h*)(pr);
            u.h[1] = *(const v8h*)(pr + 8);
            pa = u.v;
        }
#pragma unroll
        for (int t = 0; t < 4; ++t) O[t] = WMMA_F16(pa, vb[t], O[t]);
    }

    float inv[8];
#pragma unroll
    for (int j = 0; j < 8; ++j) inv[j] = lrow[j] > 0.f ? 1.f / lrow[j] : 0.f;
#pragma unroll
    for (int j = 0; j < 8; ++j) {
        int row = qtile * 16 + j + half * 8;
#pragma unroll
        for (int t = 0; t < 4; ++t)
            ctxh[(size_t)row * EMB + head * HDIM + t * 16 + col] = (_Float16)(O[t][j] * inv[j]);
    }
}

// ---------------------------------------------------------------- output projection
__global__ __launch_bounds__(128) void proj_kernel(
    const _Float16* __restrict__ ctxh, const _Float16* __restrict__ wh,
    const float* __restrict__ bias, float* __restrict__ out) {
    int wid   = blockIdx.x * 4 + (threadIdx.x >> 5);
    int mtile = wid / 16;
    int nt    = wid - mtile * 16;
    int lane  = threadIdx.x & 31;
    int col   = lane & 15;
    int half  = lane >> 4;
    int m0 = mtile * 16, n0 = nt * 64;

    v8f acc[4] = {};
    gemm_tile_16x64(ctxh + (size_t)(m0 + col) * EMB + half * 16,
                    wh   + (size_t)(n0 + col) * EMB + half * 16, acc);

#pragma unroll
    for (int j = 0; j < 8; ++j) {
        int row = m0 + j + half * 8;
        if (row >= SEQ) continue;
#pragma unroll
        for (int t = 0; t < 4; ++t) {
            int n = n0 + t * 16 + col;
            out[(size_t)row * EMB + n] = acc[t][j] + bias[n];
        }
    }
}

// ---------------------------------------------------------------- launch
extern "C" void kernel_launch(void* const* d_in, const int* in_sizes, int n_in,
                              void* d_out, int out_size, void* d_ws, size_t ws_size,
                              hipStream_t stream) {
    (void)in_sizes; (void)n_in; (void)out_size; (void)ws_size;
    const float*         x      = (const float*)d_in[0];
    const unsigned char* mask   = (const unsigned char*)d_in[1];
    const float*         qkv_w  = (const float*)d_in[2];
    const float*         qkv_b  = (const float*)d_in[3];
    const float*         proj_w = (const float*)d_in[4];
    const float*         proj_b = (const float*)d_in[5];
    const float*         cosT   = (const float*)d_in[6];
    const float*         sinT   = (const float*)d_in[7];
    float*               out    = (float*)d_out;

    char* ws = (char*)d_ws;
    size_t off = 0;
    auto alloc = [&](size_t bytes) {
        char* p = ws + off;
        off += (bytes + 255) & ~(size_t)255;
        return p;
    };
    _Float16* xh    = (_Float16*)alloc((size_t)LPAD * EMB * 2);
    _Float16* wqkv  = (_Float16*)alloc((size_t)NQKV * EMB * 2);
    _Float16* wproj = (_Float16*)alloc((size_t)EMB * EMB * 2);
    _Float16* Qh    = (_Float16*)alloc((size_t)NHEAD * LPAD * HDIM * 2);
    _Float16* Kh    = (_Float16*)alloc((size_t)NHEAD * LPAD * HDIM * 2);
    _Float16* Vt    = (_Float16*)alloc((size_t)NHEAD * HDIM * LPAD * 2);
    _Float16* ctxh  = (_Float16*)alloc((size_t)LPAD * EMB * 2);

    cvt_f32_to_f16<<<(LPAD * EMB + 255) / 256, 256, 0, stream>>>(x, xh, SEQ * EMB, LPAD * EMB);
    cvt_f32_to_f16<<<(NQKV * EMB + 255) / 256, 256, 0, stream>>>(qkv_w, wqkv, NQKV * EMB, NQKV * EMB);
    cvt_f32_to_f16<<<(EMB * EMB + 255) / 256, 256, 0, stream>>>(proj_w, wproj, EMB * EMB, EMB * EMB);

    int waves1 = (LPAD / 16) * 48;     // 7008 -> 1752 blocks of 4 waves
    qkv_rope_kernel<<<waves1 / 4, 128, 0, stream>>>(xh, wqkv, qkv_b, cosT, sinT, Qh, Kh, Vt);

    int waves2 = NHEAD * (LPAD / 16);  // 2336 -> 584 blocks
    attn_kernel<<<waves2 / 4, 128, 0, stream>>>(Qh, Kh, Vt, mask, ctxh);

    int waves3 = (LPAD / 16) * 16;     // 2336 -> 584 blocks
    proj_kernel<<<waves3 / 4, 128, 0, stream>>>(ctxh, wproj, proj_b, out);
}
